// GlobalDecoderLayer_10582799417367
// MI455X (gfx1250) — compile-verified, matrix operand order checked by reading, not run on hardware
//
#include <hip/hip_runtime.h>
#include <hip/hip_bf16.h>

// ---------------------------------------------------------------------------
// DETR-style decoder layer for MI455X (gfx1250, wave32).
//  * All GEMMs: v_wmma_f32_16x16x32_f16, f16 operands, f32 accumulate.
//  * Operands pre-converted to f16 once (halves HBM traffic vs f32).
//  * Tiles staged to LDS with gfx1250 async copies
//    (global_load_async_to_lds_b128 + s_wait_asynccnt), double-buffered so
//    the copy engine overlaps WMMA compute; uint4 register-copy fallback if
//    the builtin is unavailable.
//  * V is emitted pre-transposed by the projection GEMM epilogue so every
//    GEMM operand is k-contiguous and 16B aligned.
//  * CPB relative-position bias MLP fused (never materializes (B,NQ,NS,RPE)).
// ---------------------------------------------------------------------------

#define Bc   2
#define NQc  256
#define NSc  1024
#define Dc   256
#define Hc   8
#define DHc  32
#define FFNc 1024
#define RPEc 512
#define EPSc 1e-5f
#define SCALEc 0.17677669529663687f   // 32^-0.5

typedef __attribute__((ext_vector_type(16))) _Float16 v16h;
typedef __attribute__((ext_vector_type(8)))  float    v8f;
typedef __attribute__((ext_vector_type(4)))  int      v4i_t;

#define AS1 __attribute__((address_space(1)))
#define AS3 __attribute__((address_space(3)))

#if defined(__has_builtin)
#  if __has_builtin(__builtin_amdgcn_global_load_async_to_lds_b128)
#    define HAVE_ASYNC_LDS 1
#  endif
#endif
#ifndef HAVE_ASYNC_LDS
#  define HAVE_ASYNC_LDS 0
#endif

__device__ __forceinline__ void copy16_g2l(const _Float16* __restrict__ g,
                                           _Float16* __restrict__ l)
{
#if HAVE_ASYNC_LDS
    // Signature (from hipcc diagnostic): (int4 AS1*, int4 AS3*, imm, imm)
    __builtin_amdgcn_global_load_async_to_lds_b128(
        (AS1 v4i_t*)g, (AS3 v4i_t*)l, 0, 0);
#else
    *(uint4*)l = *(const uint4*)g;
#endif
}

__device__ __forceinline__ void wait_async0()
{
#if HAVE_ASYNC_LDS
#  if __has_builtin(__builtin_amdgcn_s_wait_asynccnt)
    __builtin_amdgcn_s_wait_asynccnt(0);
#  else
    asm volatile("s_wait_asynccnt 0x0" ::: "memory");
#  endif
#endif
}

// ---------------------------------------------------------------------------
// Batched GEMM:  acc[m,n] = alpha * sum_k A[m,k]*B[n,k]  (+bias[n])(relu)
// A,B: f16, k-contiguous rows (lda/ldb in elements), 16B-aligned rows.
// Outputs: optional f32 C (strided, + optional residual at C's index) and
// optional f16 Ch (independent strides -> free transposes).
// batch z -> (o=z/hdiv, i=z%hdiv).
// 128 threads = 4 waves; 64x64 block tile; 32x32 per wave (2x2 WMMA accums);
// K stepped by 32 through double-buffered LDS, async-staged.
// ---------------------------------------------------------------------------
__global__ __launch_bounds__(128)
void gemm_wmma_kernel(const _Float16* __restrict__ A, const _Float16* __restrict__ Bm,
                      const float* __restrict__ bias, const float* __restrict__ Res,
                      float* __restrict__ C, _Float16* __restrict__ Ch,
                      int M, int N, int K, int hdiv,
                      long a_o, long a_i, long b_o, long b_i,
                      long c_o, long c_i, long cs_m, long cs_n,
                      long ch_o, long ch_i, long chs_m, long chs_n,
                      long lda, long ldb,
                      float alpha, int do_relu)
{
    const int z = blockIdx.z;
    const int o = z / hdiv, ii = z % hdiv;
    const _Float16* __restrict__ Ab = A  + (long)o * a_o + (long)ii * a_i;
    const _Float16* __restrict__ Bb = Bm + (long)o * b_o + (long)ii * b_i;
    float*    Cb  = C  ? C  + (long)o * c_o  + (long)ii * c_i  : nullptr;
    _Float16* Chb = Ch ? Ch + (long)o * ch_o + (long)ii * ch_i : nullptr;
    const float* Rb = Res ? Res + (long)o * c_o + (long)ii * c_i : nullptr;

    const int m0 = blockIdx.y * 64;
    const int n0 = blockIdx.x * 64;

    const int tid  = threadIdx.x;
    const int lane = tid & 31;
    const int wave = tid >> 5;
    const int wr   = (wave >> 1) * 32;
    const int wc   = (wave & 1) * 32;
    const int grp  = lane >> 4;
    const int lr   = lane & 15;

    // 40-element row pitch (80B): keeps 16B chunk alignment, dodges conflicts.
    __shared__ _Float16 As[2][64][40];
    __shared__ _Float16 Bs[2][64][40];

    v8f acc[2][2] = {};

    // k offsets of the 8 half-pairs of a 16x32 f16 fragment (2 x 16B runs).
    int kmap[8];
#pragma unroll
    for (int t = 0; t < 8; ++t)
        kmap[t] = ((t >> 2) << 4) + (grp << 3) + ((t & 3) << 1);

    // Stage one 64x32 f16 tile pair: 64 rows x 4 x 16B chunks per matrix.
    auto stage = [&](int k0, int buf) {
#pragma unroll
        for (int it = 0; it < 2; ++it) {
            int idx = tid + it * 128;
            int r  = idx >> 2;
            int cq = (idx & 3) << 3;            // half-offset of 16B chunk
            int gm = m0 + r; gm = (gm < M) ? gm : (M - 1);
            int gn = n0 + r; gn = (gn < N) ? gn : (N - 1);
            copy16_g2l(Ab + (long)gm * lda + k0 + cq, &As[buf][r][cq]);
            copy16_g2l(Bb + (long)gn * ldb + k0 + cq, &Bs[buf][r][cq]);
        }
    };

    int buf = 0;
    stage(0, 0);
    for (int k0 = 0; k0 < K; k0 += 32) {
        wait_async0();
        __syncthreads();
        if (k0 + 32 < K) stage(k0 + 32, buf ^ 1);   // overlaps with WMMA below

        union Frag { v16h v; unsigned u[8]; } af[2], bf[2];
#pragma unroll
        for (int s = 0; s < 2; ++s) {
            int ar = wr + s * 16 + lr;
            int br = wc + s * 16 + lr;
#pragma unroll
            for (int t = 0; t < 8; ++t) {
                af[s].u[t] = *(const unsigned*)&As[buf][ar][kmap[t]];
                bf[s].u[t] = *(const unsigned*)&Bs[buf][br][kmap[t]];
            }
        }
#pragma unroll
        for (int si = 0; si < 2; ++si)
#pragma unroll
            for (int sj = 0; sj < 2; ++sj)
                acc[si][sj] = __builtin_amdgcn_wmma_f32_16x16x32_f16(
                    false, af[si].v, false, bf[sj].v,
                    (short)0, acc[si][sj], false, false);
        buf ^= 1;
    }

    // Epilogue. C/D fragment layout: VGPR e -> M = e + 8*grp ; N = lane&15.
#pragma unroll
    for (int si = 0; si < 2; ++si)
#pragma unroll
        for (int sj = 0; sj < 2; ++sj)
#pragma unroll
            for (int e = 0; e < 8; ++e) {
                int m = m0 + wr + si * 16 + (grp << 3) + e;
                int n = n0 + wc + sj * 16 + lr;
                if (m < M && n < N) {
                    float v = acc[si][sj][e] * alpha;
                    if (bias)    v += bias[n];
                    if (do_relu) v = fmaxf(v, 0.f);
                    if (Cb) {
                        long ci = (long)m * cs_m + (long)n * cs_n;
                        if (Rb) v += Rb[ci];
                        Cb[ci] = v;
                    }
                    if (Chb)
                        Chb[(long)m * chs_m + (long)n * chs_n] = (_Float16)v;
                }
            }
}

// ---------------------------------------------------------------------------
// Fused CPB bias MLP, added in-place into cross-attention logits (f32):
//   logits[b,h,q,k] += sum_r relu(mask[b,q,k]*w1[r] + b1[r]) * w2[h,r]
// ---------------------------------------------------------------------------
__global__ __launch_bounds__(256)
void cpb_bias_add_kernel(const float* __restrict__ mask,
                         const float* __restrict__ w1, const float* __restrict__ b1,
                         const float* __restrict__ w2, float* __restrict__ logits)
{
    __shared__ float w1s[RPEc];
    __shared__ float b1s[RPEc];
    __shared__ float w2s[Hc * RPEc];
    const int tid = threadIdx.x;
    for (int i = tid; i < RPEc; i += 256) { w1s[i] = w1[i]; b1s[i] = b1[i]; }
    for (int i = tid; i < Hc * RPEc; i += 256) w2s[i] = w2[i];
    __syncthreads();

    const long pos = (long)blockIdx.x * 256 + tid;   // over B*NQ*NS
    const long b   = pos / ((long)NQc * NSc);
    const long rem = pos % ((long)NQc * NSc);
    const long q   = rem / NSc;
    const long k   = rem % NSc;

    const float m = mask[pos];
    float accH[Hc] = {};
#pragma unroll 4
    for (int r = 0; r < RPEc; ++r) {
        float hv = fmaxf(fmaf(m, w1s[r], b1s[r]), 0.f);
#pragma unroll
        for (int h = 0; h < Hc; ++h)
            accH[h] = fmaf(hv, w2s[h * RPEc + r], accH[h]);
    }
#pragma unroll
    for (int h = 0; h < Hc; ++h) {
        long idx = (((b * Hc + h) * NQc + q) * NSc) + k;
        logits[idx] += accH[h];
    }
}

// ---------------------------------------------------------------------------
// Row softmax: reads f32 logits, writes f16 probabilities (A of the AV GEMM).
// Optional padding-mask term pad[b,k]*(-100) fused. ncols in {256,1024}.
// ---------------------------------------------------------------------------
__global__ __launch_bounds__(256)
void softmax_rows_kernel(const float* __restrict__ logits, _Float16* __restrict__ probs,
                         const float* __restrict__ pad, int ncols)
{
    const int tid = threadIdx.x;
    const long row = blockIdx.x;                 // (b*H + h)*NQ + q
    const long b   = row / ((long)Hc * NQc);
    const float* __restrict__ rp = logits + row * ncols;
    _Float16* __restrict__   pp = probs + row * ncols;

    __shared__ float red[256];
    float vals[4];
    int   cnt = 0;
    float locmax = -3.4e38f;
    for (int c = tid; c < ncols; c += 256) {
        float v = rp[c];
        if (pad) v = fmaf(pad[b * NSc + c], -100.f, v);
        vals[cnt++] = v;
        locmax = fmaxf(locmax, v);
    }
    red[tid] = locmax; __syncthreads();
    for (int s = 128; s > 0; s >>= 1) {
        if (tid < s) red[tid] = fmaxf(red[tid], red[tid + s]);
        __syncthreads();
    }
    const float mx = red[0]; __syncthreads();

    float lsum = 0.f;
    for (int i = 0; i < cnt; ++i) { vals[i] = expf(vals[i] - mx); lsum += vals[i]; }
    red[tid] = lsum; __syncthreads();
    for (int s = 128; s > 0; s >>= 1) {
        if (tid < s) red[tid] += red[tid + s];
        __syncthreads();
    }
    const float inv = 1.f / red[0]; __syncthreads();

    cnt = 0;
    for (int c = tid; c < ncols; c += 256) pp[c] = (_Float16)(vals[cnt++] * inv);
}

// ---------------------------------------------------------------------------
// LayerNorm over D=256; one 256-thread block per row; optional f16 shadow.
// ---------------------------------------------------------------------------
__global__ __launch_bounds__(256)
void layernorm_kernel(const float* __restrict__ in, float* __restrict__ out,
                      _Float16* __restrict__ outh,
                      const float* __restrict__ g, const float* __restrict__ bta)
{
    const int tid = threadIdx.x;
    const long row = blockIdx.x;
    __shared__ float red[256];

    float x = in[row * Dc + tid];
    red[tid] = x; __syncthreads();
    for (int s = 128; s > 0; s >>= 1) {
        if (tid < s) red[tid] += red[tid + s];
        __syncthreads();
    }
    const float mean = red[0] * (1.f / Dc); __syncthreads();
    float d = x - mean;
    red[tid] = d * d; __syncthreads();
    for (int s = 128; s > 0; s >>= 1) {
        if (tid < s) red[tid] += red[tid + s];
        __syncthreads();
    }
    const float var = red[0] * (1.f / Dc); __syncthreads();
    float y = d * rsqrtf(var + EPSc) * g[tid] + bta[tid];
    out[row * Dc + tid] = y;
    if (outh) outh[row * Dc + tid] = (_Float16)y;
}

__global__ __launch_bounds__(256)
void add2h_kernel(const float* __restrict__ a, const float* __restrict__ b,
                  _Float16* __restrict__ c, long n)
{
    long i = (long)blockIdx.x * 256 + threadIdx.x;
    if (i < n) c[i] = (_Float16)(a[i] + b[i]);
}

__global__ __launch_bounds__(256)
void f32_to_f16_kernel(const float* __restrict__ in, _Float16* __restrict__ out, long n)
{
    long i = (long)blockIdx.x * 256 + threadIdx.x;
    if (i < n) out[i] = (_Float16)in[i];
}

// ---------------------------------------------------------------------------
static inline void launch_gemm(hipStream_t st,
    const _Float16* A, const _Float16* Bm, const float* bias, const float* res,
    float* C, _Float16* Ch,
    int M, int N, int K, int batches, int hdiv,
    long a_o, long a_i, long b_o, long b_i,
    long c_o, long c_i, long cs_m, long cs_n,
    long ch_o, long ch_i, long chs_m, long chs_n,
    long lda, long ldb, float alpha, int relu)
{
    dim3 grid((N + 63) / 64, (M + 63) / 64, batches);
    gemm_wmma_kernel<<<grid, dim3(128), 0, st>>>(A, Bm, bias, res, C, Ch,
        M, N, K, hdiv, a_o, a_i, b_o, b_i, c_o, c_i, cs_m, cs_n,
        ch_o, ch_i, chs_m, chs_n, lda, ldb, alpha, relu);
}

extern "C" void kernel_launch(void* const* d_in, const int* in_sizes, int n_in,
                              void* d_out, int out_size, void* d_ws, size_t ws_size,
                              hipStream_t stream)
{
    const float* tgt     = (const float*)d_in[0];
    const float* qpos    = (const float*)d_in[1];
    const float* src     = (const float*)d_in[2];
    const float* spe     = (const float*)d_in[3];
    const float* padmask = (const float*)d_in[4];
    const float* cmask   = (const float*)d_in[5];
    const float* qw      = (const float*)d_in[6];
    const float* qb      = (const float*)d_in[7];
    const float* kw      = (const float*)d_in[8];
    const float* kb      = (const float*)d_in[9];
    const float* vw      = (const float*)d_in[10];
    const float* vb      = (const float*)d_in[11];
    const float* proj_w  = (const float*)d_in[12];
    const float* proj_b  = (const float*)d_in[13];
    const float* cpb_w1  = (const float*)d_in[14];
    const float* cpb_b1  = (const float*)d_in[15];
    const float* cpb_w2  = (const float*)d_in[16];
    const float* inp_w   = (const float*)d_in[17];
    const float* inp_b   = (const float*)d_in[18];
    const float* out_w   = (const float*)d_in[19];
    const float* out_b   = (const float*)d_in[20];
    const float* ln1_g   = (const float*)d_in[21];
    const float* ln1_b   = (const float*)d_in[22];
    const float* ln2_g   = (const float*)d_in[23];
    const float* ln2_b   = (const float*)d_in[24];
    const float* ln3_g   = (const float*)d_in[25];
    const float* ln3_b   = (const float*)d_in[26];
    const float* ff1_w   = (const float*)d_in[27];
    const float* ff1_b   = (const float*)d_in[28];
    const float* ff2_w   = (const float*)d_in[29];
    const float* ff2_b   = (const float*)d_in[30];
    float* out = (float*)d_out;

    // Bump allocator over d_ws (256B-aligned slices).
    char*  base = (char*)d_ws;
    size_t off  = 0;
    auto allocB = [&](size_t bytes) -> void* {
        off = (off + 255) & ~(size_t)255;
        void* p = base + off;
        off += bytes;
        return p;
    };
    auto allocF = [&](long n) { return (float*)allocB((size_t)n * 4); };
    auto allocH = [&](long n) { return (_Float16*)allocB((size_t)n * 2 + 64); };

    const long nTD = (long)Bc * NQc * Dc;   // 131072
    const long nSD = (long)Bc * NSc * Dc;   // 524288
    const long nAT = (long)Bc * Hc * NQc * NSc; // 4,194,304 (cross logits)

    // f32 scratch
    float* attn = allocF(nAT);
    float* xa   = allocF(nTD);
    float* xb   = allocF(nTD);
    float* xc   = allocF(nTD);

    // f16 scratch
    _Float16* qkh   = allocH(nTD);
    _Float16* tgth  = allocH(nTD);
    _Float16* srch  = allocH(nSD);
    _Float16* ksrch = allocH(nSD);
    _Float16* qsh   = allocH(nTD);
    _Float16* ksh   = allocH(nTD);
    _Float16* vsth  = allocH(nTD);          // self V^T  (b,h,dh,nq)
    _Float16* qch   = allocH(nTD);
    _Float16* kch   = allocH(nSD);
    _Float16* vcth  = allocH(nSD);          // cross V^T (b,h,dh,ns)
    _Float16* prbh  = allocH(nAT);          // probabilities (self then cross)
    _Float16* oth   = allocH(nTD);
    _Float16* xch   = allocH(nTD);
    _Float16* hbh   = allocH((long)Bc * NQc * FFNc);
    _Float16* inpwh = allocH(3L * Dc * Dc);
    _Float16* qwh   = allocH((long)Dc * Dc);
    _Float16* kwh   = allocH((long)Dc * Dc);
    _Float16* vwh   = allocH((long)Dc * Dc);
    _Float16* prwh  = allocH((long)Dc * Dc);
    _Float16* outwh = allocH((long)Dc * Dc);
    _Float16* ff1wh = allocH((long)FFNc * Dc);
    _Float16* ff2wh = allocH((long)Dc * FFNc);

    auto cvt = [&](const float* s, _Float16* d, long n) {
        f32_to_f16_kernel<<<(unsigned)((n + 255) / 256), 256, 0, stream>>>(s, d, n);
    };

    // ---- one-time f32 -> f16 conversions --------------------------------
    cvt(tgt, tgth, nTD);
    cvt(src, srch, nSD);
    cvt(inp_w, inpwh, 3L * Dc * Dc);
    cvt(qw, qwh, (long)Dc * Dc);
    cvt(kw, kwh, (long)Dc * Dc);
    cvt(vw, vwh, (long)Dc * Dc);
    cvt(proj_w, prwh, (long)Dc * Dc);
    cvt(out_w, outwh, (long)Dc * Dc);
    cvt(ff1_w, ff1wh, (long)FFNc * Dc);
    cvt(ff2_w, ff2wh, (long)Dc * FFNc);

    // ---- self-attention -------------------------------------------------
    add2h_kernel<<<(unsigned)((nTD + 255) / 256), 256, 0, stream>>>(tgt, qpos, qkh, nTD);

    // q,k projections -> (B*NQ, D) f16
    launch_gemm(stream, qkh, inpwh,           inp_b,       nullptr, nullptr, qsh,
        Bc*NQc, Dc, Dc, 1,1, 0,0, 0,0, 0,0,0,0, 0,0, Dc,1, Dc, Dc, 1.f, 0);
    launch_gemm(stream, qkh, inpwh + Dc*Dc,   inp_b + Dc,  nullptr, nullptr, ksh,
        Bc*NQc, Dc, Dc, 1,1, 0,0, 0,0, 0,0,0,0, 0,0, Dc,1, Dc, Dc, 1.f, 0);
    // v projection -> transposed f16 (b,h,dh,nq); batched over b.
    launch_gemm(stream, tgth, inpwh + 2*Dc*Dc, inp_b + 2*Dc, nullptr, nullptr, vsth,
        NQc, Dc, Dc, Bc,1, (long)NQc*Dc,0, 0,0, 0,0,0,0,
        (long)Hc*DHc*NQc,0, 1,NQc, Dc, Dc, 1.f, 0);

    // scores = (q k^T)*SCALE, batched (b,h) -> f32 logits
    launch_gemm(stream, qsh, ksh, nullptr, nullptr, attn, nullptr,
        NQc, NQc, DHc, Bc*Hc, Hc,
        (long)NQc*Dc, DHc, (long)NQc*Dc, DHc,
        (long)Hc*NQc*NQc, (long)NQc*NQc, NQc, 1,
        0,0,0,0, Dc, Dc, SCALEc, 0);
    softmax_rows_kernel<<<Bc*Hc*NQc, 256, 0, stream>>>(attn, prbh, nullptr, NQc);

    // context = P @ V  (B = V^T, k-contiguous) -> (B,NQ,D) f16
    launch_gemm(stream, prbh, vsth, nullptr, nullptr, nullptr, oth,
        NQc, DHc, NQc, Bc*Hc, Hc,
        (long)Hc*NQc*NQc, (long)NQc*NQc, (long)Hc*DHc*NQc, (long)DHc*NQc,
        0,0,0,0,
        (long)NQc*Dc, DHc, Dc, 1,
        NQc, NQc, 1.f, 0);

    // output projection + residual(tgt) -> xb ; LN2 -> xa
    launch_gemm(stream, oth, outwh, out_b, tgt, xb, nullptr,
        Bc*NQc, Dc, Dc, 1,1, 0,0, 0,0, 0,0, Dc,1, 0,0,0,0, Dc, Dc, 1.f, 0);
    layernorm_kernel<<<Bc*NQc, 256, 0, stream>>>(xb, xa, nullptr, ln2_g, ln2_b);

    // ---- cross-attention ------------------------------------------------
    add2h_kernel<<<(unsigned)((nTD + 255) / 256), 256, 0, stream>>>(xa, qpos, qkh, nTD);
    add2h_kernel<<<(unsigned)((nSD + 255) / 256), 256, 0, stream>>>(src, spe, ksrch, nSD);

    launch_gemm(stream, qkh, qwh, qb, nullptr, nullptr, qch,
        Bc*NQc, Dc, Dc, 1,1, 0,0, 0,0, 0,0,0,0, 0,0, Dc,1, Dc, Dc, 1.f, 0);
    launch_gemm(stream, ksrch, kwh, kb, nullptr, nullptr, kch,
        Bc*NSc, Dc, Dc, 1,1, 0,0, 0,0, 0,0,0,0, 0,0, Dc,1, Dc, Dc, 1.f, 0);
    // cross V -> transposed (b,h,dh,ns); batched over b.
    launch_gemm(stream, srch, vwh, vb, nullptr, nullptr, vcth,
        NSc, Dc, Dc, Bc,1, (long)NSc*Dc,0, 0,0, 0,0,0,0,
        (long)Hc*DHc*NSc,0, 1,NSc, Dc, Dc, 1.f, 0);

    // cross scores -> f32 logits (B,H,NQ,NS)
    launch_gemm(stream, qch, kch, nullptr, nullptr, attn, nullptr,
        NQc, NSc, DHc, Bc*Hc, Hc,
        (long)NQc*Dc, DHc, (long)NSc*Dc, DHc,
        (long)Hc*NQc*NSc, (long)NQc*NSc, NSc, 1,
        0,0,0,0, Dc, Dc, SCALEc, 0);

    // fused CPB bias MLP (+= logits), padded softmax -> f16 probs
    cpb_bias_add_kernel<<<(unsigned)((Bc*NQc*NSc) / 256), 256, 0, stream>>>(
        cmask, cpb_w1, cpb_b1, cpb_w2, attn);
    softmax_rows_kernel<<<Bc*Hc*NQc, 256, 0, stream>>>(attn, prbh, padmask, NSc);

    // context = P @ V -> (B,NQ,D) f16
    launch_gemm(stream, prbh, vcth, nullptr, nullptr, nullptr, oth,
        NQc, DHc, NSc, Bc*Hc, Hc,
        (long)Hc*NQc*NSc, (long)NQc*NSc, (long)Hc*DHc*NSc, (long)DHc*NSc,
        0,0,0,0,
        (long)NQc*Dc, DHc, Dc, 1,
        NSc, NSc, 1.f, 0);

    // projection + residual(xa) -> xb ; LN1 -> xc (+ f16 shadow)
    launch_gemm(stream, oth, prwh, proj_b, xa, xb, nullptr,
        Bc*NQc, Dc, Dc, 1,1, 0,0, 0,0, 0,0, Dc,1, 0,0,0,0, Dc, Dc, 1.f, 0);
    layernorm_kernel<<<Bc*NQc, 256, 0, stream>>>(xb, xc, xch, ln1_g, ln1_b);

    // ---- FFN ------------------------------------------------------------
    launch_gemm(stream, xch, ff1wh, ff1_b, nullptr, nullptr, hbh,
        Bc*NQc, FFNc, Dc, 1,1, 0,0, 0,0, 0,0,0,0, 0,0, FFNc,1, Dc, Dc, 1.f, 1);
    launch_gemm(stream, hbh, ff2wh, ff2_b, xc, out, nullptr,
        Bc*NQc, Dc, FFNc, 1,1, 0,0, 0,0, 0,0, Dc,1, 0,0,0,0, FFNc, FFNc, 1.f, 0);
    layernorm_kernel<<<Bc*NQc, 256, 0, stream>>>(out, out, nullptr, ln3_g, ln3_b);
}
